// TimeAttention_28252294873493
// MI455X (gfx1250) — matrix operand order repaired and checked by reading, not compiled, not run on hardware
//
#include <hip/hip_runtime.h>
#include <hip/hip_bf16.h>

// ---------------------------------------------------------------------------
// Flash attention for the "TimeAttention" reference on MI455X (gfx1250).
//   34.4 GFLOP vs ~67MB of HBM traffic -> compute-bound -> bf16 WMMA path.
//   v_wmma_f32_16x16x32_bf16 does 8x the MACs/instr of the f32 K=4 WMMA.
// ---------------------------------------------------------------------------

typedef __bf16 bf16x16 __attribute__((ext_vector_type(16)));
typedef float  f32x8   __attribute__((ext_vector_type(8)));
typedef float  f32x2   __attribute__((ext_vector_type(2)));
typedef int    i32x4   __attribute__((ext_vector_type(4)));

#define DEV __device__ __forceinline__

static constexpr int BB    = 2;
static constexpr int LL    = 2048;
static constexpr int HH    = 16;
static constexpr int EE    = 64;
static constexpr int ROTW  = EE / 2;      // 32 rotated dims
static constexpr int QTILE = 128;         // queries per block (8 waves x 16)
static constexpr int KTILE = 64;          // keys per step
static constexpr int NSTEP = LL / KTILE;  // 32
static constexpr int STR   = 72;          // padded LDS row stride (bf16 elems)
static constexpr float SCALE = 0.125f;    // 1/sqrt(64)
static constexpr float NEGBIG = -1.0e30f;

DEV unsigned short f2bf(float f) {
  union { float f; unsigned u; } x; x.f = f;
  unsigned r = x.u + 0x7FFFu + ((x.u >> 16) & 1u);  // round-to-nearest-even
  return (unsigned short)(r >> 16);
}

// two contiguous ds_load_b128 -> one 16-element bf16 WMMA fragment
DEV bf16x16 ld_frag(const unsigned short* p0, const unsigned short* p1) {
  union { i32x4 q[2]; bf16x16 v; } u;
  u.q[0] = *(const i32x4*)p0;
  u.q[1] = *(const i32x4*)p1;
  return u.v;
}

DEV void rope2(float& x, float& y, int pos, int j) {
  // theta_j = 10000^(-j/16), pair (2j, 2j+1)
  float th = __expf((float)j * (-9.210340371976184f / 16.0f));
  float ang = (float)pos * th;
  float s, c;
  __sincosf(ang, &s, &c);
  float rx = c * x - s * y;
  float ry = c * y + s * x;
  x = rx; y = ry;
}

__global__ __launch_bounds__(256)
void ta_flash_wmma(const float* __restrict__ qg,
                   const float* __restrict__ kg,
                   const float* __restrict__ vg,
                   const float* __restrict__ bwg,
                   const int*   __restrict__ ntokp,
                   float*       __restrict__ outg) {
  // LDS: K tile [key][d], V tile transposed [d][s], per-wave Q/P buffer [m][*]
  __shared__ __attribute__((aligned(16))) unsigned short sK [KTILE * STR];
  __shared__ __attribute__((aligned(16))) unsigned short sVt[EE    * STR];
  __shared__ __attribute__((aligned(16))) unsigned short sP [8 * 16 * STR];

  const int tid  = threadIdx.x;
  const int wave = tid >> 5, lane = tid & 31;
  const int nlo  = lane & 15, hi = lane >> 4;
  const int rlo  = hi * 8;                       // C/D layout: row = rlo + vgpr
  const int bb   = blockIdx.z, hh = blockIdx.y;
  const int qb   = blockIdx.x * QTILE + wave * 16;
  const int NT   = *ntokp;
  const float bw0 = bwg[hh], bw1 = bwg[HH + hh];

  unsigned short* sQw = &sP[wave * 16 * STR];    // wave-private 16x64 buffer

  const size_t rowstr = (size_t)HH * EE;         // floats between seq positions
  const float* qrow = qg + ((size_t)bb * LL) * rowstr + (size_t)hh * EE;
  const float* krow = kg + ((size_t)bb * LL) * rowstr + (size_t)hh * EE;
  const float* vrow = vg + ((size_t)bb * LL) * rowstr + (size_t)hh * EE;

  // ---- stage RoPE'd Q (bf16) into the wave buffer, pull A fragments ----
  for (int p = lane; p < 16 * (EE / 2); p += 32) {
    int m = p >> 5, j = p & 31;
    int l = qb + m;
    const float* src = qrow + (size_t)l * rowstr + 2 * j;
    f32x2 xy = *(const f32x2*)src;
    float x = xy[0], y = xy[1];
    if (j < ROTW / 2) rope2(x, y, l, j);
    sQw[m * STR + 2 * j]     = f2bf(x);
    sQw[m * STR + 2 * j + 1] = f2bf(y);
  }
  // same-wave DS ops are in-order: fragments see the stores above.
  bf16x16 qf0 = ld_frag(&sQw[nlo * STR +      16 * hi], &sQw[nlo * STR +  8 + 16 * hi]);
  bf16x16 qf1 = ld_frag(&sQw[nlo * STR + 32 + 16 * hi], &sQw[nlo * STR + 40 + 16 * hi]);

  // ---- flash-attention running state (f32) ----
  float mrow[8], lrow[8];
  int   tl[8], vl[8];
#pragma unroll
  for (int v = 0; v < 8; ++v) {
    mrow[v] = -3.0e38f; lrow[v] = 0.f;
    int l = qb + rlo + v;
    tl[v] = l % NT;  vl[v] = l / NT;
  }
  f32x8 O[4] = {};

  for (int t = 0; t < NSTEP; ++t) {
    const int sb = t * KTILE;
    __syncthreads();   // previous step's LDS readers are done

    // ---- cooperative K tile (RoPE'd) -> sK[key][d] ----
    for (int p = tid; p < KTILE * (EE / 2); p += 256) {
      int r = p >> 5, j = p & 31;
      int s = sb + r;
      const float* src = krow + (size_t)s * rowstr + 2 * j;
      f32x2 xy = *(const f32x2*)src;
      float x = xy[0], y = xy[1];
      if (j < ROTW / 2) rope2(x, y, s, j);
      sK[r * STR + 2 * j]     = f2bf(x);
      sK[r * STR + 2 * j + 1] = f2bf(y);
    }
    // ---- cooperative V tile -> sVt[d][s] (transposed for B fragments) ----
    for (int p = tid; p < KTILE * (EE / 2); p += 256) {
      int r = p >> 5, j = p & 31;
      const float* src = vrow + (size_t)(sb + r) * rowstr + 2 * j;
      f32x2 xy = *(const f32x2*)src;
      sVt[(2 * j)     * STR + r] = f2bf(xy[0]);
      sVt[(2 * j + 1) * STR + r] = f2bf(xy[1]);
    }
    __syncthreads();

    // prefetch next K/V tile toward L2 (global_prefetch_b8), overlap compute
    if (t + 1 < NSTEP) {
      int r = tid >> 2, cl = tid & 3;
      const char* kp = (const char*)(krow + (size_t)(sb + KTILE + r) * rowstr) + cl * 64;
      const char* vp = (const char*)(vrow + (size_t)(sb + KTILE + r) * rowstr) + cl * 64;
      __builtin_prefetch(kp, 0, 1);
      __builtin_prefetch(vp, 0, 1);
    }

    // ---- S = Q K^T : 4 key sub-tiles x 2 contraction chunks ----
    f32x8 S[4];
#pragma unroll
    for (int a = 0; a < 4; ++a) {
      const unsigned short* kp0 = &sK[(a * 16 + nlo) * STR +      16 * hi];
      const unsigned short* kp1 = &sK[(a * 16 + nlo) * STR + 32 + 16 * hi];
      bf16x16 bf0 = ld_frag(kp0, kp0 + 8);
      bf16x16 bf1 = ld_frag(kp1, kp1 + 8);
      f32x8 acc = {};
      acc = __builtin_amdgcn_wmma_f32_16x16x32_bf16(false, qf0, false, bf0,
                                                    (short)0, acc, false, false);
      acc = __builtin_amdgcn_wmma_f32_16x16x32_bf16(false, qf1, false, bf1,
                                                    (short)0, acc, false, false);
      S[a] = acc;
    }

    // ---- logits: scale + var-bias + within-block time mask ----
    int ts[4], vs[4];
#pragma unroll
    for (int a = 0; a < 4; ++a) {
      int s = sb + a * 16 + nlo;
      ts[a] = s % NT;  vs[a] = s / NT;
    }
    float P[4][8];
#pragma unroll
    for (int a = 0; a < 4; ++a)
#pragma unroll
      for (int v = 0; v < 8; ++v) {
        float z = S[a][v] * SCALE + ((vs[a] == vl[v]) ? bw1 : bw0);
        P[a][v] = (ts[a] > tl[v]) ? NEGBIG : z;
      }

    // ---- online softmax: row stats via half-wave shfl_xor reductions ----
    float alpha[8];
#pragma unroll
    for (int v = 0; v < 8; ++v) {
      float x = fmaxf(fmaxf(P[0][v], P[1][v]), fmaxf(P[2][v], P[3][v]));
      x = fmaxf(x, __shfl_xor(x, 1, 16));
      x = fmaxf(x, __shfl_xor(x, 2, 16));
      x = fmaxf(x, __shfl_xor(x, 4, 16));
      x = fmaxf(x, __shfl_xor(x, 8, 16));
      float mn = fmaxf(mrow[v], x);
      float al = __expf(mrow[v] - mn);
      mrow[v] = mn;
      float sum = 0.f;
#pragma unroll
      for (int a = 0; a < 4; ++a) {
        float pe = __expf(P[a][v] - mn);
        P[a][v] = pe;
        sum += pe;
      }
      sum += __shfl_xor(sum, 1, 16);
      sum += __shfl_xor(sum, 2, 16);
      sum += __shfl_xor(sum, 4, 16);
      sum += __shfl_xor(sum, 8, 16);
      lrow[v] = lrow[v] * al + sum;
      alpha[v] = al;
    }
#pragma unroll
    for (int d = 0; d < 4; ++d)
#pragma unroll
      for (int v = 0; v < 8; ++v) O[d][v] *= alpha[v];

    // ---- C/D layout -> A layout for P via wave-private LDS round-trip ----
#pragma unroll
    for (int a = 0; a < 4; ++a)
#pragma unroll
      for (int v = 0; v < 8; ++v)
        sQw[(rlo + v) * STR + a * 16 + nlo] = f2bf(P[a][v]);

    bf16x16 pf0 = ld_frag(&sQw[nlo * STR +      16 * hi], &sQw[nlo * STR +  8 + 16 * hi]);
    bf16x16 pf1 = ld_frag(&sQw[nlo * STR + 32 + 16 * hi], &sQw[nlo * STR + 40 + 16 * hi]);

    // ---- O += P V : 4 head-dim sub-tiles x 2 contraction chunks ----
#pragma unroll
    for (int d = 0; d < 4; ++d) {
      const unsigned short* vp0 = &sVt[(d * 16 + nlo) * STR +      16 * hi];
      const unsigned short* vp1 = &sVt[(d * 16 + nlo) * STR + 32 + 16 * hi];
      bf16x16 vf0 = ld_frag(vp0, vp0 + 8);
      bf16x16 vf1 = ld_frag(vp1, vp1 + 8);
      O[d] = __builtin_amdgcn_wmma_f32_16x16x32_bf16(false, pf0, false, vf0,
                                                     (short)0, O[d], false, false);
      O[d] = __builtin_amdgcn_wmma_f32_16x16x32_bf16(false, pf1, false, vf1,
                                                     (short)0, O[d], false, false);
    }
  }

  // ---- epilogue: normalize and store (B, L, H, E) f32 ----
  float* orow = outg + ((size_t)bb * LL) * rowstr + (size_t)hh * EE;
#pragma unroll
  for (int v = 0; v < 8; ++v) {
    float inv = 1.0f / lrow[v];
    int l = qb + rlo + v;
#pragma unroll
    for (int d = 0; d < 4; ++d)
      orow[(size_t)l * rowstr + d * 16 + nlo] = O[d][v] * inv;
  }
}

extern "C" void kernel_launch(void* const* d_in, const int* in_sizes, int n_in,
                              void* d_out, int out_size, void* d_ws, size_t ws_size,
                              hipStream_t stream) {
  (void)in_sizes; (void)n_in; (void)out_size; (void)d_ws; (void)ws_size;
  const float* q  = (const float*)d_in[0];
  const float* k  = (const float*)d_in[1];
  const float* v  = (const float*)d_in[2];
  const float* bw = (const float*)d_in[3];
  const int*   nt = (const int*)d_in[5];   // n_tokens (d_in[4] = n_vars unused)
  float* out = (float*)d_out;

  dim3 grid(LL / QTILE, HH, BB);  // (16, 16, 2)
  dim3 block(256);                 // 8 wave32 waves per block
  ta_flash_wmma<<<grid, block, 0, stream>>>(q, k, v, bw, nt, out);
}
